// MultiheadAttention_43671227466274
// MI455X (gfx1250) — compile-verified, matrix operand order checked by reading, not run on hardware
//
#include <hip/hip_runtime.h>

typedef __attribute__((ext_vector_type(16))) _Float16 v16h;
typedef __attribute__((ext_vector_type(8)))  _Float16 v8h;
typedef __attribute__((ext_vector_type(8)))  float    v8f;
typedef __attribute__((ext_vector_type(4)))  unsigned v4u;
typedef __attribute__((ext_vector_type(8)))  int      v8i;
typedef __attribute__((ext_vector_type(4)))  int      v4i;

#define B_   2
#define S_   2048
#define D_   1024
#define H_   16
#define DP_  64
#define MTOT (B_ * S_)

#define LDT 40   // padded LDS row stride (halves) for 32-wide tiles: 80B rows, 16B aligned
#define LKT 72   // padded LDS row stride (halves) for 64-wide rows: 144B rows, 16B aligned

// ---------------------------------------------------------------------------
// Load one 16x32 f16 WMMA operand fragment from a row-major tile.
// Per CDNA5 ISA: lane L holds row (L&15); lanes 0-15 carry
// K = {0..7, 16..23}, lanes 16-31 carry K = {8..15, 24..31}. Each half is a
// contiguous 8-element (16 B) chunk -> two b128 loads per lane.
// Works for A (row=M) and, on a transposed [N][K] tile, for B (row=N).
// ---------------------------------------------------------------------------
__device__ __forceinline__ v16h ldfrag(const _Float16* p, int ld) {
  const int lane = threadIdx.x & 31;
  const int row  = lane & 15;
  const int g    = (lane >> 4) & 1;
  const _Float16* base = p + row * ld + g * 8;
  v8h lo = *(const v8h*)(base);
  v8h hi = *(const v8h*)(base + 16);
  return __builtin_shufflevector(lo, hi, 0, 1, 2, 3, 4, 5, 6, 7,
                                         8, 9, 10, 11, 12, 13, 14, 15);
}

__device__ __forceinline__ v8f wmma_f16(v16h a, v16h b, v8f c) {
  return __builtin_amdgcn_wmma_f32_16x16x32_f16(false, a, false, b,
                                                (short)0, c, false, false);
}

// Row reductions across the 16 lanes that share a C-matrix row group.
__device__ __forceinline__ float rowmax16(float x) {
  #pragma unroll
  for (int off = 1; off < 16; off <<= 1) x = fmaxf(x, __shfl_xor(x, off, 16));
  return x;
}
__device__ __forceinline__ float rowsum16(float x) {
  #pragma unroll
  for (int off = 1; off < 16; off <<= 1) x += __shfl_xor(x, off, 16);
  return x;
}

// ---------------------------------------------------------------------------
// Tensor Data Mover: stage one 2D f16 tile (tile_w x tile_h) from global into
// LDS, with LDS row padding so the staged tile lands with a padded row stride
// matching ldfrag's 16B-aligned expectations.
// D# layout per CDNA5 ISA:
//   group0: count=1 | lds_addr | global_addr[56:0] | type=2
//   group1: data_size=1(2B), pad ctl, tensor dims, tile dims, dim0 stride
// This toolchain's builtin takes 6 args (groups 0..3 + extra group + cpol).
// Issued by one wave per block (TDM ignores EXEC; one issue per tile).
// Completion tracked with TENSORcnt.
// ---------------------------------------------------------------------------
__device__ __forceinline__ void tdm_load_tile_2d(
    void* lds_dst, const _Float16* gsrc,
    unsigned tile_w, unsigned tile_h,
    unsigned tensor_w, unsigned tensor_h, unsigned row_stride,
    unsigned pad_interval_code, unsigned pad_amount_code) {
  const unsigned long long ga = (unsigned long long)(size_t)gsrc;
  const unsigned lds_addr = (unsigned)(size_t)lds_dst;  // LDS aperture: addr[31:0]
  v4u g0;
  g0.x = 1u;                                            // count=1 (valid), user mode
  g0.y = lds_addr;
  g0.z = (unsigned)ga;                                  // global_addr[31:0]
  g0.w = (unsigned)((ga >> 32) & 0x01FFFFFFu) | (2u << 30);  // addr[56:32], type=2
  const unsigned dw0 = (1u << 16)                       // data_size = 1 -> 2 bytes
                     | (1u << 20)                       // pad_enable
                     | (pad_interval_code << 22)
                     | (pad_amount_code << 25);
  v8i g1;
  g1[0] = (int)dw0;
  g1[1] = (int)((tensor_w & 0xFFFFu) << 16);                               // dim0[15:0]
  g1[2] = (int)(((tensor_w >> 16) & 0xFFFFu) | ((tensor_h & 0xFFFFu) << 16));
  g1[3] = (int)(((tensor_h >> 16) & 0xFFFFu) | ((tile_w & 0xFFFFu) << 16));
  g1[4] = (int)(tile_h & 0xFFFFu);                                         // tile_dim2=0
  g1[5] = (int)row_stride;                                                 // dim0_stride
  g1[6] = 0;
  g1[7] = 0;
  const v4i z4 = {0, 0, 0, 0};
  const v8i z8 = {0, 0, 0, 0, 0, 0, 0, 0};
  __builtin_amdgcn_tensor_load_to_lds(g0, g1, z4, z4, z8, 0);
}

// ===========================================================================
// Kernel 1: fused QKV projections.  out = head_split(X @ W + b) as f16,
// stored [B][H][S][DP].  blockIdx.z selects which of Q/K/V to produce.
// Block = 256 threads (8 waves), tile 64(M) x 128(N), K-step 32.
// (fp32 -> f16 conversion happens on the way into LDS, so TDM does not apply;
// prefetch the next weight tile instead.)
// ===========================================================================
__global__ __launch_bounds__(256)
void qkv_proj_kernel(const float* __restrict__ Xq, const float* __restrict__ Xk,
                     const float* __restrict__ Xv,
                     const float* __restrict__ Wq, const float* __restrict__ Wk,
                     const float* __restrict__ Wv,
                     const float* __restrict__ bq, const float* __restrict__ bk,
                     const float* __restrict__ bv,
                     _Float16* __restrict__ Qh, _Float16* __restrict__ Kh,
                     _Float16* __restrict__ Vh) {
  __shared__ _Float16 As[64 * LDT];    // A tile, row-major [M][K]
  __shared__ _Float16 Bs[128 * LDT];   // B tile, transposed [N][K]

  const int t    = threadIdx.x;
  const int lane = t & 31;
  const int w    = t >> 5;
  const int m0   = blockIdx.x * 64;
  const int n0   = blockIdx.y * 128;
  const int sel  = blockIdx.z;

  const float* X    = (sel == 0) ? Xq : (sel == 1) ? Xk : Xv;
  const float* W    = (sel == 0) ? Wq : (sel == 1) ? Wk : Wv;
  const float* bias = (sel == 0) ? bq : (sel == 1) ? bk : bv;
  _Float16*    Out  = (sel == 0) ? Qh : (sel == 1) ? Kh : Vh;

  const int wy = w >> 2;   // 0..1 -> M half
  const int wx = w & 3;    // 0..3 -> N quarter

  const v8f vzero = {0.f, 0.f, 0.f, 0.f, 0.f, 0.f, 0.f, 0.f};
  v8f acc[2][2];
  acc[0][0] = vzero; acc[0][1] = vzero; acc[1][0] = vzero; acc[1][1] = vzero;

  for (int k0 = 0; k0 < D_; k0 += 32) {
    // --- stage A tile: 64x32 fp32 -> f16 (512 float4 chunks / 256 threads) ---
    #pragma unroll
    for (int i = 0; i < 2; ++i) {
      const int idx = t + 256 * i;
      const int row = idx >> 3, c4 = idx & 7;
      const float4 f = *(const float4*)(X + (size_t)(m0 + row) * D_ + k0 + c4 * 4);
      _Float16* dst = &As[row * LDT + c4 * 4];
      dst[0] = (_Float16)f.x; dst[1] = (_Float16)f.y;
      dst[2] = (_Float16)f.z; dst[3] = (_Float16)f.w;
    }
    // --- stage B tile transposed: W[k0..+31][n0..+127] -> Bs[N][K] ---
    #pragma unroll
    for (int i = 0; i < 4; ++i) {
      const int idx = t + 256 * i;
      const int row = idx >> 5, c4 = idx & 31;
      const float4 f = *(const float4*)(W + (size_t)(k0 + row) * D_ + n0 + c4 * 4);
      Bs[(c4 * 4 + 0) * LDT + row] = (_Float16)f.x;
      Bs[(c4 * 4 + 1) * LDT + row] = (_Float16)f.y;
      Bs[(c4 * 4 + 2) * LDT + row] = (_Float16)f.z;
      Bs[(c4 * 4 + 3) * LDT + row] = (_Float16)f.w;
    }
    // prefetch next K-step's weight tile while this step computes
    if (k0 + 32 < D_)
      __builtin_prefetch(W + (size_t)(k0 + 32 + (t >> 5)) * D_ + n0 + (t & 31) * 4, 0, 0);
    __syncthreads();

    v16h a0 = ldfrag(&As[(wy * 32 +  0) * LDT], LDT);
    v16h a1 = ldfrag(&As[(wy * 32 + 16) * LDT], LDT);
    v16h b0 = ldfrag(&Bs[(wx * 32 +  0) * LDT], LDT);
    v16h b1 = ldfrag(&Bs[(wx * 32 + 16) * LDT], LDT);

    acc[0][0] = wmma_f16(a0, b0, acc[0][0]);
    acc[0][1] = wmma_f16(a0, b1, acc[0][1]);
    acc[1][0] = wmma_f16(a1, b0, acc[1][0]);
    acc[1][1] = wmma_f16(a1, b1, acc[1][1]);
    __syncthreads();
  }

  // Epilogue: + bias, convert to f16, scatter into head-split [B][H][S][DP].
  const int g = (lane >> 4) & 1, cn = lane & 15;
  #pragma unroll
  for (int mi = 0; mi < 2; ++mi)
    #pragma unroll
    for (int ni = 0; ni < 2; ++ni)
      #pragma unroll
      for (int r = 0; r < 8; ++r) {
        const int gm = m0 + wy * 32 + mi * 16 + r + 8 * g;
        const int gn = n0 + wx * 32 + ni * 16 + cn;
        const float v = acc[mi][ni][r] + bias[gn];
        const int bb = gm >> 11;        // / S_
        const int s  = gm & (S_ - 1);
        const int h  = gn >> 6;         // / DP_
        const int d  = gn & (DP_ - 1);
        Out[(((size_t)bb * H_ + h) * S_ + s) * DP_ + d] = (_Float16)v;
      }
}

// ===========================================================================
// Kernel 2: causal flash attention per (b,h).  Block = 128 threads (4 waves);
// block owns 64 query rows, each wave owns 16.  K tiles staged via the Tensor
// Data Mover (DMA with LDS row padding 64->72 halves), V transposed manually;
// online softmax state (m, l) lives in VGPRs; logits never touch HBM.
// ===========================================================================
__global__ __launch_bounds__(128)
void flash_attn_kernel(const _Float16* __restrict__ Qh,
                       const _Float16* __restrict__ Kh,
                       const _Float16* __restrict__ Vh,
                       _Float16* __restrict__ Ctx) {
  __shared__ _Float16 Ks[64 * LKT];     // K tile, row-major [key][d]   (B of Q@K^T)
  __shared__ _Float16 Vt[64 * LKT];     // V tile, transposed [d][key]  (B of P@V)
  __shared__ _Float16 Ps[4][16 * 40];   // per-wave P staging (C-layout -> A-layout)

  const int t    = threadIdx.x;
  const int lane = t & 31;
  const int w    = t >> 5;
  const int qb   = blockIdx.x * 64;
  const int bh   = blockIdx.y;                 // b*H + h
  const int g    = (lane >> 4) & 1;
  const int cn   = lane & 15;

  const _Float16* Qbase = Qh + (size_t)bh * S_ * DP_;
  const _Float16* Kbase = Kh + (size_t)bh * S_ * DP_;
  const _Float16* Vbase = Vh + (size_t)bh * S_ * DP_;

  const int qrow0 = qb + w * 16;
  // Q fragments held in registers for the whole kernel (d = 0..31, 32..63).
  v16h qa0 = ldfrag(Qbase + (size_t)qrow0 * DP_ +  0, DP_);
  v16h qa1 = ldfrag(Qbase + (size_t)qrow0 * DP_ + 32, DP_);

  const v8f vzero = {0.f, 0.f, 0.f, 0.f, 0.f, 0.f, 0.f, 0.f};
  v8f O[4];
  O[0] = vzero; O[1] = vzero; O[2] = vzero; O[3] = vzero;
  float m[8], l[8];
  #pragma unroll
  for (int r = 0; r < 8; ++r) { m[r] = -1e30f; l[r] = 0.f; }

  const float scl = 0.125f;  // 1/sqrt(DP_)

  for (int kv = 0; kv <= qb + 63; kv += 64) {
    // --- K tile via TDM DMA: 64(d) x 64(keys), LDS rows padded to 72 halves
    //     (row = 64 halves = 32 DW -> pad_interval code 4; pad 4 DW -> code 3)
    if (w == 0)
      tdm_load_tile_2d(&Ks[0], Kbase + (size_t)kv * DP_,
                       /*tile*/ DP_, 64, /*tensor*/ DP_, S_, /*stride*/ DP_,
                       /*interval*/ 4, /*amount*/ 3);
    // --- V tile transposed by all waves (overlaps with the DMA) ---
    #pragma unroll
    for (int i = 0; i < 4; ++i) {
      const int idx = t + 128 * i;          // 512 chunks of 8 halves
      const int row = idx >> 3, c8 = idx & 7;
      v8h vv = *(const v8h*)(Vbase + (size_t)(kv + row) * DP_ + c8 * 8);
      #pragma unroll
      for (int e = 0; e < 8; ++e) Vt[(c8 * 8 + e) * LKT + row] = vv[e];
    }
    if (w == 0) __builtin_amdgcn_s_wait_tensorcnt(0);
    __syncthreads();

    #pragma unroll
    for (int kc = 0; kc < 64; kc += 32) {
      // ---- logits S = Q @ K^T for 16 q-rows x 32 keys ----
      v8f s0 = vzero, s1 = vzero;
      v16h bk0 = ldfrag(&Ks[(kc +  0) * LKT +  0], LKT);
      v16h bk1 = ldfrag(&Ks[(kc +  0) * LKT + 32], LKT);
      v16h bk2 = ldfrag(&Ks[(kc + 16) * LKT +  0], LKT);
      v16h bk3 = ldfrag(&Ks[(kc + 16) * LKT + 32], LKT);
      s0 = wmma_f16(qa0, bk0, s0);
      s0 = wmma_f16(qa1, bk1, s0);
      s1 = wmma_f16(qa0, bk2, s1);
      s1 = wmma_f16(qa1, bk3, s1);

      // ---- causal mask + online softmax update ----
      float corr[8];
      #pragma unroll
      for (int r = 0; r < 8; ++r) {
        const int qi  = qrow0 + r + 8 * g;
        const int kj0 = kv + kc + cn;
        const int kj1 = kj0 + 16;
        const float x0 = (kj0 <= qi) ? s0[r] * scl : -1e9f;
        const float x1 = (kj1 <= qi) ? s1[r] * scl : -1e9f;
        const float rm = rowmax16(fmaxf(x0, x1));
        const float nm = fmaxf(m[r], rm);
        const float c  = __expf(m[r] - nm);
        m[r] = nm;
        const float p0 = __expf(x0 - nm);
        const float p1 = __expf(x1 - nm);
        l[r] = l[r] * c + rowsum16(p0 + p1);
        corr[r] = c;
        const int prow = r + 8 * g;
        Ps[w][prow * 40 + cn]      = (_Float16)p0;
        Ps[w][prow * 40 + 16 + cn] = (_Float16)p1;
      }

      // rescale running output by correction factor (row-consistent layout)
      #pragma unroll
      for (int ni = 0; ni < 4; ++ni)
        #pragma unroll
        for (int r = 0; r < 8; ++r) O[ni][r] *= corr[r];

      // ---- O += P @ V  (A-frag of P via wave-private LDS transpose) ----
      v16h pa = ldfrag(&Ps[w][0], 40);
      #pragma unroll
      for (int ni = 0; ni < 4; ++ni) {
        v16h bv = ldfrag(&Vt[(ni * 16) * LKT + kc], LKT);
        O[ni] = wmma_f16(pa, bv, O[ni]);
      }
    }
    __syncthreads();
  }

  // ---- finalize: divide by l, write ctx in concat layout [B*S][H*DP] ----
  float inv[8];
  #pragma unroll
  for (int r = 0; r < 8; ++r) inv[r] = 1.0f / l[r];
  const int b = bh >> 4, h = bh & (H_ - 1);
  #pragma unroll
  for (int ni = 0; ni < 4; ++ni)
    #pragma unroll
    for (int r = 0; r < 8; ++r) {
      const int srow = qrow0 + r + 8 * g;
      const int dcol = ni * 16 + cn;
      Ctx[((size_t)b * S_ + srow) * D_ + h * DP_ + dcol] =
          (_Float16)(O[ni][r] * inv[r]);
    }
}

// ===========================================================================
// Kernel 3: output projection  out = ctx @ Wo + bo  (fp32 result).
// A operand (ctx, already f16) is staged with TDM (LDS rows padded 32->40
// halves); weight tile converted fp32->f16 by the waves.
// ===========================================================================
__global__ __launch_bounds__(256)
void out_proj_kernel(const _Float16* __restrict__ Ctx,
                     const float* __restrict__ Wo,
                     const float* __restrict__ bo,
                     float* __restrict__ Out) {
  __shared__ _Float16 As[64 * LDT];
  __shared__ _Float16 Bs[128 * LDT];

  const int t    = threadIdx.x;
  const int lane = t & 31;
  const int w    = t >> 5;
  const int m0   = blockIdx.x * 64;
  const int n0   = blockIdx.y * 128;
  const int wy   = w >> 2, wx = w & 3;

  const v8f vzero = {0.f, 0.f, 0.f, 0.f, 0.f, 0.f, 0.f, 0.f};
  v8f acc[2][2];
  acc[0][0] = vzero; acc[0][1] = vzero; acc[1][0] = vzero; acc[1][1] = vzero;

  for (int k0 = 0; k0 < D_; k0 += 32) {
    // A tile via TDM: 32(k) x 64(m) f16 tile, row = 32 halves = 16 DW ->
    // pad_interval code 3; pad 4 DW -> code 3  => LDS stride 40 halves.
    if (w == 0)
      tdm_load_tile_2d(&As[0], Ctx + (size_t)m0 * D_ + k0,
                       /*tile*/ 32, 64, /*tensor*/ D_, MTOT, /*stride*/ D_,
                       /*interval*/ 3, /*amount*/ 3);
    #pragma unroll
    for (int i = 0; i < 4; ++i) {  // B tile transposed, fp32 -> f16
      const int idx = t + 256 * i;
      const int row = idx >> 5, c4 = idx & 31;
      const float4 f = *(const float4*)(Wo + (size_t)(k0 + row) * D_ + n0 + c4 * 4);
      Bs[(c4 * 4 + 0) * LDT + row] = (_Float16)f.x;
      Bs[(c4 * 4 + 1) * LDT + row] = (_Float16)f.y;
      Bs[(c4 * 4 + 2) * LDT + row] = (_Float16)f.z;
      Bs[(c4 * 4 + 3) * LDT + row] = (_Float16)f.w;
    }
    if (k0 + 32 < D_)
      __builtin_prefetch(Wo + (size_t)(k0 + 32 + (t >> 5)) * D_ + n0 + (t & 31) * 4, 0, 0);
    if (w == 0) __builtin_amdgcn_s_wait_tensorcnt(0);
    __syncthreads();

    v16h a0 = ldfrag(&As[(wy * 32 +  0) * LDT], LDT);
    v16h a1 = ldfrag(&As[(wy * 32 + 16) * LDT], LDT);
    v16h b0 = ldfrag(&Bs[(wx * 32 +  0) * LDT], LDT);
    v16h b1 = ldfrag(&Bs[(wx * 32 + 16) * LDT], LDT);

    acc[0][0] = wmma_f16(a0, b0, acc[0][0]);
    acc[0][1] = wmma_f16(a0, b1, acc[0][1]);
    acc[1][0] = wmma_f16(a1, b0, acc[1][0]);
    acc[1][1] = wmma_f16(a1, b1, acc[1][1]);
    __syncthreads();
  }

  const int g = (lane >> 4) & 1, cn = lane & 15;
  #pragma unroll
  for (int mi = 0; mi < 2; ++mi)
    #pragma unroll
    for (int ni = 0; ni < 2; ++ni)
      #pragma unroll
      for (int r = 0; r < 8; ++r) {
        const int gm = m0 + wy * 32 + mi * 16 + r + 8 * g;
        const int gn = n0 + wx * 32 + ni * 16 + cn;
        Out[(size_t)gm * D_ + gn] = acc[mi][ni][r] + bo[gn];
      }
}

// ===========================================================================
extern "C" void kernel_launch(void* const* d_in, const int* in_sizes, int n_in,
                              void* d_out, int out_size, void* d_ws, size_t ws_size,
                              hipStream_t stream) {
  (void)in_sizes; (void)n_in; (void)out_size; (void)ws_size;
  const float* q  = (const float*)d_in[0];
  const float* k  = (const float*)d_in[1];
  const float* v  = (const float*)d_in[2];
  // d_in[3] = causal mask -> computed analytically in-kernel (saves 16.8 MB reads)
  const float* Wq = (const float*)d_in[4];
  const float* Wk = (const float*)d_in[5];
  const float* Wv = (const float*)d_in[6];
  const float* Wo = (const float*)d_in[7];
  const float* bq = (const float*)d_in[8];
  const float* bk = (const float*)d_in[9];
  const float* bv = (const float*)d_in[10];
  const float* bo = (const float*)d_in[11];
  float* out = (float*)d_out;

  const size_t per = (size_t)B_ * H_ * S_ * DP_;  // elements per Q/K/V buffer
  _Float16* Qh  = (_Float16*)d_ws;
  _Float16* Kh  = Qh + per;
  _Float16* Vh  = Kh + per;
  _Float16* Ctx = Vh + per;   // total workspace: 4 * 8 MB = 33.5 MB

  dim3 g1(MTOT / 64, D_ / 128, 3);
  qkv_proj_kernel<<<g1, 256, 0, stream>>>(q, k, v, Wq, Wk, Wv, bq, bk, bv,
                                          Qh, Kh, Vh);
  dim3 g2(S_ / 64, B_ * H_);
  flash_attn_kernel<<<g2, 128, 0, stream>>>(Qh, Kh, Vh, Ctx);

  dim3 g3(MTOT / 64, D_ / 128);
  out_proj_kernel<<<g3, 256, 0, stream>>>(Ctx, Wo, bo, out);
}
